// GraphEncoder_4904852652248
// MI455X (gfx1250) — matrix operand order repaired and checked by reading, not compiled, not run on hardware
//
#include <hip/hip_runtime.h>
#include <hip/hip_bf16.h>
#include <math.h>

#define IN_DIM   128
#define HID_DIM  64
#define OUT_DIM  64
#define NCOLS    192   // [basis0 out | basis1 out | self-loop out]

typedef __attribute__((ext_vector_type(16))) __bf16 v16bf;
typedef __attribute__((ext_vector_type(8)))  float  v8f;

__device__ __forceinline__ __bf16 f2bf(float x) {
  return (__bf16)x;   // RNE fptrunc; lowers to v_cvt_pk_bf16_f32 pairs
}

// Pack basis[2, K, 64] and loop_w[K, 64] (f32) into column-major bf16 Wb[192][K].
__global__ void pack_w_kernel(const float* __restrict__ basis,
                              const float* __restrict__ loop_w,
                              __bf16* __restrict__ Wb, int K) {
  int idx = blockIdx.x * blockDim.x + threadIdx.x;
  int total = NCOLS * K;
  if (idx >= total) return;
  int col = idx / K;
  int k   = idx - col * K;
  float v;
  if (col < 128) {
    int b = col >> 6, o = col & 63;
    v = basis[((size_t)b * K + k) * 64 + o];
  } else {
    v = loop_w[(size_t)k * 64 + (col - 128)];
  }
  Wb[(size_t)col * K + k] = f2bf(v);
}

// C[Npad,192] = A[N,K] (f32, cast to bf16) @ W[K,192], W packed col-major bf16.
// One wave32 per 16-row stripe; 4 waves per block; 12 column tiles per wave.
// C has Npad = roundup(N,16) rows, so stores are unconditional.
template<int K>
__global__ void gemm_wmma_bf16(const float* __restrict__ A,
                               const __bf16* __restrict__ Bm,  // [192][K] col-major
                               float* __restrict__ C,
                               int N) {
  constexpr int KSTEPS = K / 32;
  const int lane = threadIdx.x & 31;
  const int wave = threadIdx.x >> 5;
  const int row0 = (blockIdx.x * 4 + wave) * 16;
  if (row0 >= N) return;                 // wave-uniform exit; EXEC stays all-ones
  const int r  = lane & 15;
  const int kh = lane >> 4;              // which K half this lane carries

  int arow = row0 + r; if (arow >= N) arow = N - 1;   // clamp input-side loads
  const float* ap = A + (size_t)arow * K;

  // A fragment per k-step (contiguous k order used here, same as B):
  // lanes 0-15: K = s*32 + {0..15}; lanes 16-31: K = s*32 + {16..31}
  v16bf afrag[KSTEPS];
#pragma unroll
  for (int s = 0; s < KSTEPS; ++s) {
    const float* p = ap + s * 32 + kh * 16;
    float4 f0 = *(const float4*)(p);
    float4 f1 = *(const float4*)(p + 4);
    float4 f2 = *(const float4*)(p + 8);
    float4 f3 = *(const float4*)(p + 12);
    v16bf a;
    a[0]=f2bf(f0.x);  a[1]=f2bf(f0.y);  a[2]=f2bf(f0.z);  a[3]=f2bf(f0.w);
    a[4]=f2bf(f1.x);  a[5]=f2bf(f1.y);  a[6]=f2bf(f1.z);  a[7]=f2bf(f1.w);
    a[8]=f2bf(f2.x);  a[9]=f2bf(f2.y);  a[10]=f2bf(f2.z); a[11]=f2bf(f2.w);
    a[12]=f2bf(f3.x); a[13]=f2bf(f3.y); a[14]=f2bf(f3.z); a[15]=f2bf(f3.w);
    afrag[s] = a;
  }

  // Limit unroll to 2 column tiles: two independent WMMA chains for latency
  // hiding without keeping 12 accumulators live (which forced scratch spills).
#pragma unroll 2
  for (int ct = 0; ct < NCOLS / 16; ++ct) {
    v8f acc = {};
    // B in the same contiguous k order as A: lanes 0-15 hold K=0..15 of column n=r (contiguous),
    // lanes 16-31 hold K=16..31 -> one contiguous 32B load from the packed column.
    const __bf16* bp = Bm + (size_t)(ct * 16 + r) * K + kh * 16;
#pragma unroll
    for (int s = 0; s < KSTEPS; ++s) {
      v16bf b = *(const v16bf*)(bp + s * 32);
      acc = __builtin_amdgcn_wmma_f32_16x16x32_bf16(
                false, afrag[s], false, b, (short)0, acc, false, false);
    }
    // C/D layout: lane col = r, rows = row0 + 8*kh + i. One base address,
    // per-store compile-time byte offsets; C is row-padded so no guards needed.
    const int ncol = ct * 16 + r;
    float* cbase = C + (size_t)(row0 + kh * 8) * NCOLS + ncol;
#pragma unroll
    for (int i = 0; i < 8; ++i) cbase[(size_t)i * NCOLS] = acc[i];
  }
}

// One wave32 per edge: msg = c0*hb[src, 0:64] + c1*hb[src, 64:128]; scatter to agg[dst].
__global__ void edge_scatter(const float* __restrict__ hb,
                             const float* __restrict__ comp,  // [R,2]
                             const int* __restrict__ src,
                             const int* __restrict__ dst,
                             const int* __restrict__ et,
                             float* __restrict__ agg, int E) {
  int t = blockIdx.x * blockDim.x + threadIdx.x;
  // e is the same for all 32 lanes of a wave; make it provably uniform so the
  // per-edge metadata loads become scalar (SMEM) loads.
  int e = __builtin_amdgcn_readfirstlane(t >> 5);
  if (e >= E) return;
  int lane = t & 31;
  int sn = src[e], dn = dst[e], rr = et[e];
  float c0 = comp[rr * 2 + 0];
  float c1 = comp[rr * 2 + 1];
  const float2* h0 = (const float2*)(hb + (size_t)sn * NCOLS);
  float2 a = h0[lane];           // cols 2*lane, 2*lane+1       (basis 0)
  float2 b = h0[32 + lane];      // cols 64+2*lane, 64+2*lane+1 (basis 1)
  float m0 = fmaf(c1, b.x, c0 * a.x);
  float m1 = fmaf(c1, b.y, c0 * a.y);
  float* o = agg + (size_t)dn * 64 + lane * 2;
  unsafeAtomicAdd(o,     m0);    // hardware global_atomic_add_f32
  unsafeAtomicAdd(o + 1, m1);
}

// out[n,c] = act(agg[n,c] + hb[n,128+c] + bias[c]);  mode 0 = tanh, 1 = relu
__global__ void finalize_act(const float* __restrict__ agg,
                             const float* __restrict__ hb,
                             const float* __restrict__ bias,
                             float* __restrict__ out, int total, int mode) {
  int i = blockIdx.x * blockDim.x + threadIdx.x;
  if (i >= total) return;
  int c = i & 63;
  int n = i >> 6;
  float x = agg[i] + hb[(size_t)n * NCOLS + 128 + c] + bias[c];
  out[i] = (mode == 0) ? tanhf(x) : fmaxf(x, 0.0f);
}

extern "C" void kernel_launch(void* const* d_in, const int* in_sizes, int n_in,
                              void* d_out, int out_size, void* d_ws, size_t ws_size,
                              hipStream_t stream) {
  const float* node_emb = (const float*)d_in[0];
  const float* basis1   = (const float*)d_in[1];
  const float* comp1    = (const float*)d_in[2];
  const float* loop_w1  = (const float*)d_in[3];
  const float* bias1    = (const float*)d_in[4];
  const float* basis2   = (const float*)d_in[5];
  const float* comp2    = (const float*)d_in[6];
  const float* loop_w2  = (const float*)d_in[7];
  const float* bias2    = (const float*)d_in[8];
  const int*   src      = (const int*)d_in[9];
  const int*   dst      = (const int*)d_in[10];
  const int*   et       = (const int*)d_in[11];

  const int N = in_sizes[0] / IN_DIM;
  const int E = in_sizes[9];
  const int Npad = (N + 15) & ~15;       // padded rows for unguarded WMMA stores

  // workspace carve-up (~128 MB)
  float*  hb  = (float*)d_ws;                        // Npad*192 f32 (both layers)
  float*  agg = hb  + (size_t)Npad * NCOLS;          // N*64    f32
  float*  h1  = agg + (size_t)N * 64;                // Npad*64 f32 (layer-1 out)
  __bf16* Wb1 = (__bf16*)(h1 + (size_t)Npad * 64);   // 192*128 bf16
  __bf16* Wb2 = Wb1 + NCOLS * IN_DIM;                // 192*64  bf16

  const int gemmBlocks = (N + 63) / 64;
  const long long edgeThreads = (long long)E * 32;
  const unsigned edgeBlocks = (unsigned)((edgeThreads + 255) / 256);
  const int total = N * OUT_DIM;

  // ---- layer 1 ----
  pack_w_kernel<<<(NCOLS * IN_DIM + 255) / 256, 256, 0, stream>>>(basis1, loop_w1, Wb1, IN_DIM);
  gemm_wmma_bf16<IN_DIM><<<gemmBlocks, 128, 0, stream>>>(node_emb, Wb1, hb, N);
  hipMemsetAsync(agg, 0, (size_t)N * 64 * sizeof(float), stream);
  edge_scatter<<<edgeBlocks, 256, 0, stream>>>(hb, comp1, src, dst, et, agg, E);
  finalize_act<<<(total + 255) / 256, 256, 0, stream>>>(agg, hb, bias1, h1, total, 0);

  // ---- layer 2 ----
  pack_w_kernel<<<(NCOLS * HID_DIM + 255) / 256, 256, 0, stream>>>(basis2, loop_w2, Wb2, HID_DIM);
  gemm_wmma_bf16<HID_DIM><<<gemmBlocks, 128, 0, stream>>>(h1, Wb2, hb, N);
  hipMemsetAsync(agg, 0, (size_t)N * 64 * sizeof(float), stream);
  edge_scatter<<<edgeBlocks, 256, 0, stream>>>(hb, comp2, src, dst, et, agg, E);
  finalize_act<<<(total + 255) / 256, 256, 0, stream>>>(agg, hb, bias2, (float*)d_out, total, 1);
}
